// AttentionBlock_15728170238395
// MI455X (gfx1250) — matrix operand order repaired and checked by reading, not compiled, not run on hardware
//
#include <hip/hip_runtime.h>
#include <hip/hip_bf16.h>

// ---------------------------------------------------------------------------
// AttentionBlock for MI455X (gfx1250, wave32, WMMA f16 16x16x32, f32 accum)
// B=16, C=512, H=W=32 (T=1024), heads=8 (ch=64), groups=32
// LDS tiles use padded row strides (multiple of 8 halves, non power-of-two)
// so 16-B ds_load_b128 fragment reads spread across the 64 LDS banks.
// V is staged via the Tensor Data Mover (tensor_load_to_lds + TENSORcnt).
// ---------------------------------------------------------------------------

typedef _Float16 half_t;
typedef __attribute__((ext_vector_type(16))) _Float16 v16h;
typedef __attribute__((ext_vector_type(8)))  _Float16 v8h;
typedef __attribute__((ext_vector_type(8)))  float    v8f;
typedef __attribute__((ext_vector_type(4)))  unsigned int u32x4;
typedef __attribute__((ext_vector_type(4)))  int i32x4;
typedef __attribute__((ext_vector_type(8)))  int i32x8;

#define CC   512
#define TT   1024
#define BB   16
#define OC   1536          // 3*C
#define NH   8
#define CH   64            // C / NH
#define HEADBLK 192        // 3*CH rows per head in qkv
#define P_SCALE 0.125f     // (64^-1/4)^2 folded into q once

// padded LDS strides (halves)
#define BSTR   520         // GEMM B tile rows of 512
#define KSTR   72          // attention K^T rows of 64
#define VCHUNK 520         // V: each 512-half chunk padded to 520 (TDM pad)
#define PSTR   40          // P re-layout rows of 32
#define QSTR   72          // q^T rows of 64

static __device__ __forceinline__ v8f wmma_f16(v16h a, v16h b, v8f c) {
    return __builtin_amdgcn_wmma_f32_16x16x32_f16(false, a, false, b, (short)0, c,
                                                  false, false);
}

static __device__ __forceinline__ v16h frag_from2(const half_t* p0, const half_t* p1) {
    v8h lo = *(const v8h*)p0;
    v8h hi = *(const v8h*)p1;
    v16h r;
    #pragma unroll
    for (int i = 0; i < 8; ++i) { r[i] = lo[i]; r[i + 8] = hi[i]; }
    return r;
}

// ---------------------------------------------------------------------------
// GroupNorm stats: one block per (b, g) — contiguous 16384-element slice.
// ---------------------------------------------------------------------------
__global__ void __launch_bounds__(256)
attnblk_gn_stats(const float* __restrict__ x, float* __restrict__ stats) {
    __shared__ float ssum[256];
    __shared__ float ssq[256];
    const int bg = blockIdx.x;
    const float* p = x + (size_t)bg * 16384;
    float s = 0.f, q = 0.f;
    for (int i = threadIdx.x; i < 16384; i += 256) {
        float v = p[i];
        s += v; q += v * v;
    }
    ssum[threadIdx.x] = s; ssq[threadIdx.x] = q;
    __syncthreads();
    for (int off = 128; off; off >>= 1) {
        if ((int)threadIdx.x < off) {
            ssum[threadIdx.x] += ssum[threadIdx.x + off];
            ssq[threadIdx.x]  += ssq[threadIdx.x + off];
        }
        __syncthreads();
    }
    if (threadIdx.x == 0) {
        float mean = ssum[0] * (1.f / 16384.f);
        float var  = ssq[0] * (1.f / 16384.f) - mean * mean;
        stats[bg * 2 + 0] = mean;
        stats[bg * 2 + 1] = rsqrtf(var + 1e-5f);
    }
}

__global__ void __launch_bounds__(256)
attnblk_gn_apply(const float* __restrict__ x, const float* __restrict__ stats,
                 const float* __restrict__ gw, const float* __restrict__ gb,
                 half_t* __restrict__ xn) {
    size_t i = (size_t)blockIdx.x * 256 + threadIdx.x;
    int c  = (int)((i >> 10) & (CC - 1));
    int bg = (int)(i >> 14);
    float mean = stats[bg * 2 + 0];
    float rstd = stats[bg * 2 + 1];
    float v = (x[i] - mean) * rstd * gw[c] + gb[c];
    xn[i] = (half_t)v;
}

__global__ void __launch_bounds__(256)
attnblk_cast_f16(const float* __restrict__ in, half_t* __restrict__ out, int n) {
    int i = blockIdx.x * 256 + threadIdx.x;
    if (i < n) out[i] = (half_t)in[i];
}

// ---------------------------------------------------------------------------
// QKV GEMM: out[b,o,t] = sum_c W[o,c]*xn[b,c,t] + bias[o]  (f16 out).
// ---------------------------------------------------------------------------
__global__ void __launch_bounds__(256)
attnblk_gemm_qkv(const half_t* __restrict__ W, const half_t* __restrict__ Xn,
                 const float* __restrict__ bias, half_t* __restrict__ Out) {
    extern __shared__ char dynsmem[];
    half_t* smem = (half_t*)dynsmem;                  // [32][BSTR]
    const int b  = blockIdx.x;
    const int t0 = blockIdx.y * 32;
    const half_t* xb = Xn + (size_t)b * CC * TT;

    for (int idx = threadIdx.x; idx < 32 * CC; idx += 256) {
        int tl = idx & 31, c = idx >> 5;
        smem[tl * BSTR + c] = xb[(size_t)c * TT + t0 + tl];
    }
    __syncthreads();

    const int lane  = threadIdx.x & 31;
    const int wave  = threadIdx.x >> 5;
    const int mlane = lane & 15;
    const int half_id = lane >> 4;
    const int kbA = 8 * half_id;
    const int kbB = 16 * half_id;

    for (int mt = wave; mt < OC / 16; mt += 8) {      // 96 M-tiles, 12 per wave
        const half_t* wrow = W + (size_t)(mt * 16 + mlane) * CC;
        if (mt + 8 < OC / 16)
            __builtin_prefetch(W + (size_t)((mt + 8) * 16 + mlane) * CC, 0, 1);
        v8f acc0 = {}; v8f acc1 = {};
        #pragma unroll 4
        for (int k0 = 0; k0 < CC; k0 += 32) {
            v16h a  = frag_from2(wrow + k0 + kbB, wrow + k0 + kbB + 8);
            const half_t* bp0 = smem + (size_t)mlane * BSTR + k0 + kbB;
            const half_t* bp1 = smem + (size_t)(16 + mlane) * BSTR + k0 + kbB;
            v16h b0 = frag_from2(bp0, bp0 + 8);
            v16h b1 = frag_from2(bp1, bp1 + 8);
            acc0 = wmma_f16(a, b0, acc0);
            acc1 = wmma_f16(a, b1, acc1);
        }
        #pragma unroll
        for (int r = 0; r < 8; ++r) {
            int row = mt * 16 + r + 8 * half_id;
            float bv = bias[row];
            size_t o = ((size_t)b * OC + row) * TT + t0 + mlane;
            Out[o]      = (half_t)(acc0[r] + bv);
            Out[o + 16] = (half_t)(acc1[r] + bv);
        }
    }
}

// ---------------------------------------------------------------------------
// Attention: one workgroup per (b, head). K^T + V staged in 320 KB WGP LDS;
// V via Tensor Data Mover with LDS padding (4 DWORDs every 256 DWORDs ->
// 512-half chunks at stride VCHUNK=520). Flash-style online softmax.
// ---------------------------------------------------------------------------
__global__ void __launch_bounds__(256)
attnblk_attention(const half_t* __restrict__ QKV, half_t* __restrict__ Aout) {
    extern __shared__ char dynsmem[];
    half_t* kT = (half_t*)dynsmem;                    // [1024][KSTR]
    half_t* vS = kT + 1024 * KSTR;                    // 128 chunks * [VCHUNK]
    half_t* pS = vS + 128 * VCHUNK;                   // 8 waves * [16][PSTR]
    half_t* qT = pS + 8 * 16 * PSTR;                  // 8 waves * [16][QSTR]

    const int bh = blockIdx.x;
    const int b = bh >> 3, h = bh & 7;
    const half_t* base = QKV + (size_t)b * OC * TT + (size_t)h * HEADBLK * TT;
    const half_t* Q = base;
    const half_t* K = base + (size_t)CH * TT;
    const half_t* V = base + (size_t)2 * CH * TT;
    const int tid = threadIdx.x;
    const int lane = tid & 31, wave = tid >> 5;

    // ---- Stage V [64][1024] -> padded chunks via TDM (wave 0 issues) -------
#if defined(__has_builtin) && __has_builtin(__builtin_amdgcn_tensor_load_to_lds)
    if (wave == 0) {
        unsigned long long ga = (unsigned long long)(const void*)V;
        unsigned int lds_off = (unsigned int)(1024u * KSTR * sizeof(half_t));
        u32x4 g0;
        g0[0] = 1u;                                       // count=1 (valid D#)
        g0[1] = lds_off;                                  // lds_addr (bytes)
        g0[2] = (unsigned int)(ga & 0xFFFFFFFFu);         // global_addr lo
        g0[3] = (unsigned int)((ga >> 32) & 0x01FFFFFFu)  // global_addr hi
              | (2u << 30);                               // type=2 (image)
        i32x8 g1;
        // data_size=8B | pad_enable | pad_interval=256DW | pad_amount=4DW
        g1[0] = (int)((3u << 16) | (1u << 20) | (7u << 22) | (3u << 25));
        g1[1] = (int)(0x4000u << 16);   // tensor_dim0 = 16384 (8B units), lo16
        g1[2] = (int)(1u << 16);        // tensor_dim0 hi = 0; tensor_dim1 = 1
        g1[3] = (int)(0x4000u << 16);   // tensor_dim1 hi = 0; tile_dim0 = 16384
        g1[4] = 1;                      // tile_dim1 = 1; tile_dim2 = 0
        g1[5] = 16384;                  // tensor_dim0_stride lo32
        g1[6] = (int)(0x4000u << 16);   // stride hi = 0; tensor_dim1_stride lo16
        g1[7] = 0;
        i32x4 gz = {0, 0, 0, 0};
#if __clang_major__ >= 23
        i32x8 gz8 = {0, 0, 0, 0, 0, 0, 0, 0};
        __builtin_amdgcn_tensor_load_to_lds(g0, g1, gz, gz, gz8, 0);
#else
        __builtin_amdgcn_tensor_load_to_lds(g0, g1, gz, gz, 0);
#endif
#if defined(__has_builtin) && __has_builtin(__builtin_amdgcn_s_wait_tensorcnt)
        __builtin_amdgcn_s_wait_tensorcnt(0);
#else
        asm volatile("s_wait_tensorcnt 0x0" ::: "memory");
#endif
    }
#else
    for (int idx = tid; idx < (CH * TT) / 8; idx += 256) {
        int s8 = (idx & 127) * 8, c = idx >> 7;
        *(v8h*)(vS + ((size_t)c * 2 + (s8 >> 9)) * VCHUNK + (s8 & 511)) =
            *(const v8h*)(V + (size_t)c * TT + s8);
    }
#endif

    // ---- Stage K transposed (coalesced reads, padded-stride writes) --------
    for (int idx = tid; idx < CH * TT; idx += 256) {
        int s = idx & (TT - 1), c = idx >> 10;
        kT[s * KSTR + c] = K[(size_t)c * TT + s];
    }
    __syncthreads();

    const int nlane = lane & 15;
    const int half_id = lane >> 4;
    const int kbA = 8 * half_id;
    const int kbB = 16 * half_id;
    half_t* qTw = qT + wave * 16 * QSTR;              // [16 t][QSTR]
    half_t* pW  = pS + wave * 16 * PSTR;              // [16 m][PSTR]

    for (int qt = wave; qt < TT / 16; qt += 8) {
        const int t0 = qt * 16;

        __builtin_amdgcn_wave_barrier();
        // q tile transposed, with softmax product scale (1/8) folded in once.
        for (int idx = lane; idx < 16 * CH; idx += 32) {
            int c = idx & (CH - 1), tl = idx >> 6;
            qTw[tl * QSTR + c] =
                (half_t)((float)Q[(size_t)c * TT + t0 + tl] * P_SCALE);
        }
        __builtin_amdgcn_wave_barrier();

        v16h qa[2];
        #pragma unroll
        for (int kk = 0; kk < 2; ++kk) {
            const half_t* ap = qTw + nlane * QSTR + kk * 32 + kbB;
            qa[kk] = frag_from2(ap, ap + 8);
        }

        v8f O[4] = {{0}, {0}, {0}, {0}};
        float mrun[8], lrun[8];
        #pragma unroll
        for (int r = 0; r < 8; ++r) { mrun[r] = -3.0e38f; lrun[r] = 0.f; }

        for (int s0 = 0; s0 < TT; s0 += 32) {
            // ---- S = q^T k for 32 keys ----
            v8f S0 = {}, S1 = {};
            #pragma unroll
            for (int kk = 0; kk < 2; ++kk) {
                const half_t* bp0 = kT + (size_t)(s0 + nlane) * KSTR + kk * 32 + kbB;
                const half_t* bp1 = kT + (size_t)(s0 + 16 + nlane) * KSTR + kk * 32 + kbB;
                v16h bf0 = frag_from2(bp0, bp0 + 8);
                v16h bf1 = frag_from2(bp1, bp1 + 8);
                S0 = wmma_f16(qa[kk], bf0, S0);
                S1 = wmma_f16(qa[kk], bf1, S1);
            }

            // ---- online softmax (rows in half-wave lane groups) ----
            float alpha[8];
            #pragma unroll
            for (int r = 0; r < 8; ++r) {
                float tmax = fmaxf(S0[r], S1[r]);
                #pragma unroll
                for (int off = 8; off; off >>= 1)
                    tmax = fmaxf(tmax, __shfl_xor(tmax, off, 16));
                float mnew = fmaxf(mrun[r], tmax);
                float a = __expf(mrun[r] - mnew);
                mrun[r] = mnew;
                float p0 = __expf(S0[r] - mnew);
                float p1 = __expf(S1[r] - mnew);
                S0[r] = p0; S1[r] = p1;
                float rsum = p0 + p1;
                #pragma unroll
                for (int off = 8; off; off >>= 1)
                    rsum += __shfl_xor(rsum, off, 16);
                lrun[r] = lrun[r] * a + rsum;
                alpha[r] = a;
            }
            #pragma unroll
            for (int j = 0; j < 4; ++j)
                #pragma unroll
                for (int r = 0; r < 8; ++r) O[j][r] *= alpha[r];

            // ---- C-frag -> A-frag re-layout of P via per-wave LDS ----
            #pragma unroll
            for (int r = 0; r < 8; ++r) {
                int m = r + 8 * half_id;
                pW[m * PSTR + nlane]      = (half_t)S0[r];
                pW[m * PSTR + 16 + nlane] = (half_t)S1[r];
            }
            __builtin_amdgcn_wave_barrier();
            const half_t* ap = pW + nlane * PSTR + kbB;
            v16h pa = frag_from2(ap, ap + 8);

            // ---- O += P * V  (chunked V layout; frag never straddles) ----
            const int sv   = s0 + kbB;
            const int sblk = sv >> 9;
            const int soff = sv & 511;
            #pragma unroll
            for (int j = 0; j < 4; ++j) {
                const half_t* bp =
                    vS + (size_t)((j * 16 + nlane) * 2 + sblk) * VCHUNK + soff;
                v16h bf = frag_from2(bp, bp + 8);
                O[j] = wmma_f16(pa, bf, O[j]);
            }
        }

        // ---- epilogue: normalize, transpose through LDS, 16-B stores ----
        half_t* oW = qTw;                             // reuse: 64 c rows x 16 t
        #pragma unroll
        for (int r = 0; r < 8; ++r) lrun[r] = 1.0f / lrun[r];
        #pragma unroll
        for (int j = 0; j < 4; ++j) {
            int c = j * 16 + nlane;
            #pragma unroll
            for (int r = 0; r < 8; ++r)
                oW[c * 16 + r + 8 * half_id] = (half_t)(O[j][r] * lrun[r]);
        }
        __builtin_amdgcn_wave_barrier();
        #pragma unroll
        for (int i = 0; i < 4; ++i) {
            int chunk = lane + 32 * i;                // 128 chunks of 8 halves
            int c = chunk >> 1;
            int off = (chunk & 1) * 8;
            v8h d = *(const v8h*)(oW + c * 16 + off);
            *(v8h*)(Aout + ((size_t)b * CC + h * CH + c) * TT + t0 + off) = d;
        }
    }
}

// ---------------------------------------------------------------------------
// proj GEMM + residual: out[b,c,t] = x[b,c,t] + sum_k W[c,k]*a[b,k,t] + bias[c]
// ---------------------------------------------------------------------------
__global__ void __launch_bounds__(256)
attnblk_gemm_proj(const half_t* __restrict__ W, const half_t* __restrict__ Ain,
                  const float* __restrict__ bias, const float* __restrict__ Xres,
                  float* __restrict__ Out) {
    extern __shared__ char dynsmem[];
    half_t* smem = (half_t*)dynsmem;                  // [32][BSTR]
    const int b  = blockIdx.x;
    const int t0 = blockIdx.y * 32;
    const half_t* ab = Ain + (size_t)b * CC * TT;

    for (int idx = threadIdx.x; idx < 32 * CC; idx += 256) {
        int tl = idx & 31, c = idx >> 5;
        smem[tl * BSTR + c] = ab[(size_t)c * TT + t0 + tl];
    }
    __syncthreads();

    const int lane = threadIdx.x & 31, wave = threadIdx.x >> 5;
    const int mlane = lane & 15;
    const int half_id = lane >> 4;
    const int kbA = 8 * half_id;
    const int kbB = 16 * half_id;

    for (int mt = wave; mt < CC / 16; mt += 8) {      // 32 M-tiles, 4 per wave
        const half_t* wrow = W + (size_t)(mt * 16 + mlane) * CC;
        if (mt + 8 < CC / 16)
            __builtin_prefetch(W + (size_t)((mt + 8) * 16 + mlane) * CC, 0, 1);
        v8f acc0 = {}; v8f acc1 = {};
        #pragma unroll 4
        for (int k0 = 0; k0 < CC; k0 += 32) {
            v16h a  = frag_from2(wrow + k0 + kbB, wrow + k0 + kbB + 8);
            const half_t* bp0 = smem + (size_t)mlane * BSTR + k0 + kbB;
            const half_t* bp1 = smem + (size_t)(16 + mlane) * BSTR + k0 + kbB;
            v16h b0 = frag_from2(bp0, bp0 + 8);
            v16h b1 = frag_from2(bp1, bp1 + 8);
            acc0 = wmma_f16(a, b0, acc0);
            acc1 = wmma_f16(a, b1, acc1);
        }
        #pragma unroll
        for (int r = 0; r < 8; ++r) {
            int row = mt * 16 + r + 8 * half_id;
            float bv = bias[row];
            size_t o = ((size_t)b * CC + row) * TT + t0 + mlane;
            Out[o]      = Xres[o]      + acc0[r] + bv;
            Out[o + 16] = Xres[o + 16] + acc1[r] + bv;
        }
    }
}

// ---------------------------------------------------------------------------
// Host side
// ---------------------------------------------------------------------------
extern "C" void kernel_launch(void* const* d_in, const int* in_sizes, int n_in,
                              void* d_out, int out_size, void* d_ws, size_t ws_size,
                              hipStream_t stream) {
    (void)in_sizes; (void)n_in; (void)out_size; (void)ws_size;
    const float* x      = (const float*)d_in[0];
    const float* gn_w   = (const float*)d_in[1];
    const float* gn_b   = (const float*)d_in[2];
    const float* qkv_w  = (const float*)d_in[3];
    const float* qkv_b  = (const float*)d_in[4];
    const float* proj_w = (const float*)d_in[5];
    const float* proj_b = (const float*)d_in[6];
    float* out = (float*)d_out;

    char* ws = (char*)d_ws;
    float*  stats  = (float*)ws;                              // 512*2 f32
    half_t* wq     = (half_t*)(ws + 4096);                    // 1536*512 f16
    half_t* wp     = wq + (size_t)OC * CC;                    // 512*512 f16
    half_t* xn     = wp + (size_t)CC * CC;                    // B*C*T f16
    half_t* qkv    = xn + (size_t)BB * CC * TT;               // B*3C*T f16
    half_t* attn_a = qkv + (size_t)BB * OC * TT;              // B*C*T f16

    attnblk_gn_stats<<<BB * 32, 256, 0, stream>>>(x, stats);
    attnblk_cast_f16<<<(OC * CC) / 256, 256, 0, stream>>>(qkv_w, wq, OC * CC);
    attnblk_cast_f16<<<(CC * CC) / 256, 256, 0, stream>>>(proj_w, wp, CC * CC);
    attnblk_gn_apply<<<(BB * CC * TT) / 256, 256, 0, stream>>>(x, stats, gn_w, gn_b, xn);

    dim3 gq(BB, TT / 32);
    attnblk_gemm_qkv<<<gq, 256, 32 * BSTR * sizeof(half_t), stream>>>(wq, xn, qkv_b, qkv);

    size_t attn_lds = (size_t)(1024 * KSTR + 128 * VCHUNK + 8 * 16 * PSTR
                               + 8 * 16 * QSTR) * sizeof(half_t);  // 309248 B
    attnblk_attention<<<BB * NH, 256, attn_lds, stream>>>(qkv, attn_a);

    dim3 gp(BB, TT / 32);
    attnblk_gemm_proj<<<gp, 256, 32 * BSTR * sizeof(half_t), stream>>>(wp, attn_a, proj_b, x, out);
}
